// NeuralClustering_55473797595480
// MI455X (gfx1250) — compile-verified, matrix operand order checked by reading, not run on hardware
//
#include <hip/hip_runtime.h>
#include <math.h>

// ---------------------------------------------------------------------------
// NeuralClustering fused MLP pipeline for gfx1250 (wave32, WMMA 16x16x32 f16)
//
// Stages (all on `stream`):
//   1) 17x nc_pack_layer : f32 weights/bias -> f16, WMMA-B tiled layout
//   2) nc_zero_f32       : zero Q accumulator
//   3) nc_h_mlp          : 102400x(2->512->512->512->512->256), epilogue
//                          scatters h0/hn and atomically accumulates Q
//   4) nc_build_gin      : g_in = [h0+hn ; h0 ; hn]  (3072x256)
//   5) nc_g_mlp          : 3072x(256->512 x6)
//   6) nc_build_uu       : uu = [Gk | Q | hn]        (2048x1024)
//   7) nc_f_mlp          : 2048x(1024->512->...->1) + fused log_softmax
//
// Inner GEMM register-blocks 4 N-tiles per A-fragment load: one LDS A load
// feeds 4 consecutive v_wmma ops (independent accumulators), amortizing LDS
// bandwidth and covering the WMMA->VALU RAW hazard window.
// ---------------------------------------------------------------------------

typedef _Float16 h16;
typedef __attribute__((ext_vector_type(8)))  _Float16 v8h;
typedef __attribute__((ext_vector_type(16))) _Float16 v16h;
typedef __attribute__((ext_vector_type(8)))  float    v8f;

#define NC_MTILE   64
#define NC_THREADS 256
#define NC_NWAVES  8
#define NC_B       1024
#define NC_N       100
#define NC_LDSBYTES (2 * NC_MTILE * 512 * 2)   // two 64x512 f16 ping-pong buffers

struct NcLayers {
    const h16* W[6];
    const h16* B[6];
};

__device__ __forceinline__ v16h nc_cat8(v8h a, v8h b) {
    return __builtin_shufflevector(a, b, 0, 1, 2, 3, 4, 5, 6, 7,
                                        8, 9, 10, 11, 12, 13, 14, 15);
}

// One dense layer over a 64-row tile: Out = act(A @ W + bias)
// A:  NC_MTILE x Kpad (f16, row-major, lda elems, LDS or global)
// Wp: packed B fragments, tile (kt,nt) at ((kt*numNT)+nt)*512, lane*16 contig.
// Out: NC_MTILE x Npad (f16, row-major, ldo elems, LDS)
// NBLK consecutive N-tiles are computed per A-fragment load.
template<int NBLK>
__device__ __forceinline__ void nc_mlp_layer(
    const h16* __restrict__ A, int lda,
    const h16* __restrict__ Wp, const h16* __restrict__ Bp,
    h16* __restrict__ Out, int ldo,
    int Kpad, int Npad, bool prelu)
{
    const int lane  = threadIdx.x & 31;
    const int wave  = threadIdx.x >> 5;
    const int half  = lane >> 4;
    const int l15   = lane & 15;
    const int numNT = Npad >> 4;
    const int numKT = Kpad >> 5;
    const int numMT = NC_MTILE >> 4;
    const int numNB = numNT / NBLK;

    for (int t = wave; t < numMT * numNB; t += NC_NWAVES) {
        const int mt  = t / numNB;
        const int nb  = t - mt * numNB;
        const int nt0 = nb * NBLK;

        v8f c[NBLK];
#pragma unroll
        for (int j = 0; j < NBLK; ++j) {
            const float bias = (float)Bp[(nt0 + j) * 16 + l15];
            c[j] = (v8f){ bias, bias, bias, bias, bias, bias, bias, bias };
        }

        // A fragment rows: lane 0-15 -> M=l15 with K halves {0..7,16..23},
        //                  lane 16-31 -> M=l15 with K halves {8..15,24..31}
        const h16* arow = A + (size_t)(mt * 16 + l15) * lda + half * 8;

        for (int kt = 0; kt < numKT; ++kt) {
            v8h a0 = *(const v8h*)(arow + kt * 32);
            v8h a1 = *(const v8h*)(arow + kt * 32 + 16);
            v16h a = nc_cat8(a0, a1);

            const h16* bbase = Wp + ((size_t)(kt * numNT + nt0)) * 512 + lane * 16;
#pragma unroll
            for (int j = 0; j < NBLK; ++j) {
                v8h b0 = *(const v8h*)(bbase + (size_t)j * 512);
                v8h b1 = *(const v8h*)(bbase + (size_t)j * 512 + 8);
                v16h b = nc_cat8(b0, b1);
                c[j] = __builtin_amdgcn_wmma_f32_16x16x32_f16(
                           false, a, false, b, (short)0, c[j], false, false);
            }
        }

        // C/D layout: lanes 0-15 N=l15 M=v, lanes 16-31 N=l15 M=8+v
#pragma unroll
        for (int j = 0; j < NBLK; ++j) {
            const int n = (nt0 + j) * 16 + l15;
#pragma unroll
            for (int v = 0; v < 8; ++v) {
                const int m = mt * 16 + half * 8 + v;
                float r = c[j][v];
                if (prelu) r = fmaxf(r, 0.0f) + 0.25f * fminf(r, 0.0f);
                Out[(size_t)m * ldo + n] = (h16)r;
            }
        }
    }
}

// ---------------------------------------------------------------------------
// Weight/bias packing: f32 -> f16 in WMMA B-fragment tile order.
// ---------------------------------------------------------------------------
__global__ void nc_pack_layer(const float* __restrict__ W,
                              const float* __restrict__ bias,
                              h16* __restrict__ Wdst, h16* __restrict__ Bdst,
                              int K, int N, int Kpad, int Npad)
{
    const int numNT = Npad >> 4;
    const int total = (Kpad >> 5) * numNT * 512;
    for (int idx = blockIdx.x * blockDim.x + threadIdx.x;
         idx < total + Npad; idx += gridDim.x * blockDim.x) {
        if (idx < total) {
            const int tile = idx >> 9;
            const int r    = idx & 511;
            const int l    = r >> 4;
            const int i    = r & 15;
            const int kt   = tile / numNT;
            const int nt   = tile - kt * numNT;
            const int half = l >> 4;
            const int l15  = l & 15;
            const int k = kt * 32 + (i < 8 ? half * 8 + i : 16 + half * 8 + (i - 8));
            const int n = nt * 16 + l15;
            float v = (k < K && n < N) ? W[(size_t)k * N + n] : 0.0f;
            Wdst[idx] = (h16)v;
        } else {
            const int n = idx - total;
            float v = (bias != nullptr && n < N) ? bias[n] : 0.0f;
            Bdst[n] = (h16)v;
        }
    }
}

__global__ void nc_zero_f32(float* __restrict__ p, int n) {
    const int idx = blockIdx.x * blockDim.x + threadIdx.x;
    if (idx < n) p[idx] = 0.0f;
}

// ---------------------------------------------------------------------------
// h-MLP: rows = B*N = 102400, dims 2->512->512->512->512->256 (PReLU x4)
// Epilogue scatters n==0 -> h0, n==1 -> hn, n>=2 -> atomic Q accumulation.
// ---------------------------------------------------------------------------
__global__ void __launch_bounds__(NC_THREADS)
nc_h_mlp(const float* __restrict__ data, NcLayers L,
         h16* __restrict__ h0buf, h16* __restrict__ hnbuf,
         float* __restrict__ Q)
{
    extern __shared__ char smem[];
    h16* bufA = (h16*)smem;
    h16* bufB = bufA + NC_MTILE * 512;
    const int rowBase = blockIdx.x * NC_MTILE;

    // Stage input: 64 rows x 2 features, zero-padded to K=32.
    for (int idx = threadIdx.x; idx < NC_MTILE * 32; idx += NC_THREADS) {
        const int r = idx >> 5, c = idx & 31;
        float v = (c < 2) ? data[(size_t)(rowBase + r) * 2 + c] : 0.0f;
        bufA[idx] = (h16)v;
    }
    __syncthreads();

    nc_mlp_layer<4>(bufA, 32,  L.W[0], L.B[0], bufB, 512, 32,  512, true);  __syncthreads();
    nc_mlp_layer<4>(bufB, 512, L.W[1], L.B[1], bufA, 512, 512, 512, true);  __syncthreads();
    nc_mlp_layer<4>(bufA, 512, L.W[2], L.B[2], bufB, 512, 512, 512, true);  __syncthreads();
    nc_mlp_layer<4>(bufB, 512, L.W[3], L.B[3], bufA, 512, 512, 512, true);  __syncthreads();
    nc_mlp_layer<4>(bufA, 512, L.W[4], L.B[4], bufB, 256, 512, 256, false); __syncthreads();

    for (int idx = threadIdx.x; idx < NC_MTILE * 256; idx += NC_THREADS) {
        const int r = idx >> 8, c = idx & 255;
        const int g = rowBase + r;
        const int b = g / NC_N;
        const int n = g - b * NC_N;
        const h16 hv = bufB[idx];
        if (n == 0)      h0buf[(size_t)b * 256 + c] = hv;
        else if (n == 1) hnbuf[(size_t)b * 256 + c] = hv;
        else             atomicAdd(&Q[(size_t)b * 256 + c], (float)hv);
    }
}

// g_in = [h0+hn ; h0 ; hn] : 3072 x 256
__global__ void nc_build_gin(const h16* __restrict__ h0, const h16* __restrict__ hn,
                             h16* __restrict__ gin)
{
    const int idx = blockIdx.x * blockDim.x + threadIdx.x;
    if (idx >= 3 * NC_B * 256) return;
    const int row = idx >> 8, c = idx & 255;
    const int sel = row >> 10;       // 0: h0+hn, 1: h0, 2: hn
    const int b   = row & (NC_B - 1);
    const float a = (float)h0[(size_t)b * 256 + c];
    const float d = (float)hn[(size_t)b * 256 + c];
    const float v = (sel == 0) ? (a + d) : ((sel == 1) ? a : d);
    gin[idx] = (h16)v;
}

// g-MLP: 3072 rows, dims 256->512->512->512->512->512->512 (PReLU x5)
__global__ void __launch_bounds__(NC_THREADS)
nc_g_mlp(const h16* __restrict__ gin, NcLayers L, h16* __restrict__ gout)
{
    extern __shared__ char smem[];
    h16* bufA = (h16*)smem;
    h16* bufB = bufA + NC_MTILE * 512;
    const int rowBase = blockIdx.x * NC_MTILE;

    for (int idx = threadIdx.x; idx < NC_MTILE * 256; idx += NC_THREADS)
        bufA[idx] = gin[(size_t)rowBase * 256 + idx];
    __syncthreads();

    nc_mlp_layer<4>(bufA, 256, L.W[0], L.B[0], bufB, 512, 256, 512, true);  __syncthreads();
    nc_mlp_layer<4>(bufB, 512, L.W[1], L.B[1], bufA, 512, 512, 512, true);  __syncthreads();
    nc_mlp_layer<4>(bufA, 512, L.W[2], L.B[2], bufB, 512, 512, 512, true);  __syncthreads();
    nc_mlp_layer<4>(bufB, 512, L.W[3], L.B[3], bufA, 512, 512, 512, true);  __syncthreads();
    nc_mlp_layer<4>(bufA, 512, L.W[4], L.B[4], bufB, 512, 512, 512, true);  __syncthreads();
    nc_mlp_layer<4>(bufB, 512, L.W[5], L.B[5], bufA, 512, 512, 512, false); __syncthreads();

    for (int idx = threadIdx.x; idx < NC_MTILE * 512; idx += NC_THREADS)
        gout[(size_t)rowBase * 512 + idx] = bufA[idx];
}

// uu row (2b+k) = [Gk(b,k) 512 | Q(b) 256 | hn(b) 256] : 2048 x 1024
__global__ void nc_build_uu(const h16* __restrict__ gout, const float* __restrict__ Q,
                            const h16* __restrict__ hn, h16* __restrict__ uu)
{
    const int idx = blockIdx.x * blockDim.x + threadIdx.x;
    if (idx >= 2 * NC_B * 1024) return;
    const int row = idx >> 10, f = idx & 1023;
    const int b = row >> 1, k = row & 1;
    h16 v;
    if (f < 512) {
        if (k == 0) v = gout[(size_t)b * 512 + f];
        else        v = (h16)((float)gout[(size_t)(NC_B + b) * 512 + f] +
                              (float)gout[(size_t)(2 * NC_B + b) * 512 + f]);
    } else if (f < 768) {
        v = (h16)Q[(size_t)b * 256 + (f - 512)];
    } else {
        v = hn[(size_t)b * 256 + (f - 768)];
    }
    uu[idx] = v;
}

// f-MLP: 2048 rows, dims 1024->512->512->512->512->512->1 (PReLU x5, no last
// bias). First layer streams A from global uu; tail does pairwise log_softmax.
__global__ void __launch_bounds__(NC_THREADS)
nc_f_mlp(const h16* __restrict__ uu, NcLayers L, float* __restrict__ out)
{
    extern __shared__ char smem[];
    h16* bufA = (h16*)smem;
    h16* bufB = bufA + NC_MTILE * 512;
    const int rowBase = blockIdx.x * NC_MTILE;

    nc_mlp_layer<4>(uu + (size_t)rowBase * 1024, 1024,
                    L.W[0], L.B[0], bufA, 512, 1024, 512, true);            __syncthreads();
    nc_mlp_layer<4>(bufA, 512, L.W[1], L.B[1], bufB, 512, 512, 512, true);  __syncthreads();
    nc_mlp_layer<4>(bufB, 512, L.W[2], L.B[2], bufA, 512, 512, 512, true);  __syncthreads();
    nc_mlp_layer<4>(bufA, 512, L.W[3], L.B[3], bufB, 512, 512, 512, true);  __syncthreads();
    nc_mlp_layer<4>(bufB, 512, L.W[4], L.B[4], bufA, 512, 512, 512, true);  __syncthreads();
    nc_mlp_layer<1>(bufA, 512, L.W[5], L.B[5], bufB, 16,  512, 16,  false); __syncthreads();

    if (threadIdx.x < NC_MTILE / 2) {
        const int i = threadIdx.x;
        const float l0 = (float)bufB[(size_t)(2 * i) * 16];
        const float l1 = (float)bufB[(size_t)(2 * i + 1) * 16];
        const float m = fmaxf(l0, l1);
        const float lse = m + logf(expf(l0 - m) + expf(l1 - m));
        const int b = (rowBase >> 1) + i;
        out[b * 2 + 0] = l0 - lse;
        out[b * 2 + 1] = l1 - lse;
    }
}

// ---------------------------------------------------------------------------
// Host side
// ---------------------------------------------------------------------------
extern "C" void kernel_launch(void* const* d_in, const int* in_sizes, int n_in,
                              void* d_out, int out_size, void* d_ws, size_t ws_size,
                              hipStream_t stream)
{
    (void)in_sizes; (void)n_in; (void)out_size; (void)ws_size;

    const float* data = (const float*)d_in[0];
    char* ws = (char*)d_ws;
    size_t cur = 0;
    auto take = [&](size_t bytes) -> size_t {
        size_t o = cur;
        cur += (bytes + 15) & ~(size_t)15;
        return o;
    };

    struct PackInfo { int K, N; };
    static const PackInfo hL[5] = {{2,512},{512,512},{512,512},{512,512},{512,256}};
    static const PackInfo gL[6] = {{256,512},{512,512},{512,512},{512,512},{512,512},{512,512}};
    static const PackInfo fL[6] = {{1024,512},{512,512},{512,512},{512,512},{512,512},{512,1}};

    // Input index map (pytree flatten, None bias dropped):
    //  0 data, 1 cs, 2 n
    //  3..7   hW0..4   8..12 hb0..4   13..16 h alphas (0.25)
    //  17..22 gW0..5   23..28 gb0..5  29..33 g alphas
    //  34..39 fW0..5   40..44 fb0..4  45..49 f alphas
    NcLayers hP{}, gP{}, fP{};
    auto packMlp = [&](const PackInfo* Ls, int nL, int wBase, int bBase, int nBias,
                       NcLayers& P) {
        for (int i = 0; i < nL; ++i) {
            const int K = Ls[i].K, N = Ls[i].N;
            const int Kpad = (K + 31) & ~31;
            const int Npad = (N + 15) & ~15;
            h16* wdst = (h16*)(ws + take((size_t)Kpad * Npad * sizeof(h16)));
            h16* bdst = (h16*)(ws + take((size_t)Npad * sizeof(h16)));
            const float* wsrc = (const float*)d_in[wBase + i];
            const float* bsrc = (i < nBias) ? (const float*)d_in[bBase + i] : nullptr;
            const int elems = (Kpad >> 5) * (Npad >> 4) * 512 + Npad;
            nc_pack_layer<<<(elems + 255) / 256, 256, 0, stream>>>(
                wsrc, bsrc, wdst, bdst, K, N, Kpad, Npad);
            P.W[i] = wdst;
            P.B[i] = bdst;
        }
    };
    packMlp(hL, 5, 3, 8, 5, hP);
    packMlp(gL, 6, 17, 23, 6, gP);
    packMlp(fL, 6, 34, 40, 5, fP);

    h16*   h0buf = (h16*)(ws + take((size_t)NC_B * 256 * sizeof(h16)));
    h16*   hnbuf = (h16*)(ws + take((size_t)NC_B * 256 * sizeof(h16)));
    h16*   gin   = (h16*)(ws + take((size_t)3 * NC_B * 256 * sizeof(h16)));
    h16*   gout  = (h16*)(ws + take((size_t)3 * NC_B * 512 * sizeof(h16)));
    h16*   uu    = (h16*)(ws + take((size_t)2 * NC_B * 1024 * sizeof(h16)));
    float* Q     = (float*)(ws + take((size_t)NC_B * 256 * sizeof(float)));

    nc_zero_f32<<<(NC_B * 256 + 255) / 256, 256, 0, stream>>>(Q, NC_B * 256);

    const int hRows = NC_B * NC_N;                 // 102400
    nc_h_mlp<<<hRows / NC_MTILE, NC_THREADS, NC_LDSBYTES, stream>>>(
        data, hP, h0buf, hnbuf, Q);

    nc_build_gin<<<(3 * NC_B * 256 + 255) / 256, 256, 0, stream>>>(h0buf, hnbuf, gin);

    nc_g_mlp<<<(3 * NC_B) / NC_MTILE, NC_THREADS, NC_LDSBYTES, stream>>>(gin, gP, gout);

    nc_build_uu<<<(2 * NC_B * 1024 + 255) / 256, 256, 0, stream>>>(gout, Q, hnbuf, uu);

    nc_f_mlp<<<(2 * NC_B) / NC_MTILE, NC_THREADS, NC_LDSBYTES, stream>>>(
        uu, fP, (float*)d_out);
}
